// BiVision_VQA2_82068235092490
// MI455X (gfx1250) — compile-verified, matrix operand order checked by reading, not run on hardware
//
#include <hip/hip_runtime.h>
#include <hip/hip_bf16.h>

typedef unsigned short ushort_t;
typedef __bf16 bf16_t;
typedef __attribute__((ext_vector_type(16))) __bf16 v16bf;
typedef __attribute__((ext_vector_type(8)))  float  v8f;
typedef __attribute__((__vector_size__(16))) int v4i_b;

#if __has_builtin(__builtin_amdgcn_global_load_async_to_lds_b128) && \
    __has_builtin(__builtin_amdgcn_s_wait_asynccnt)
#define USE_ASYNC_LDS 1
#else
#define USE_ASYNC_LDS 0
#endif

#define GLOBAL_AS __attribute__((address_space(1)))
#define LDS_AS    __attribute__((address_space(3)))

// ---------- bf16 helpers (manual RNE) ----------
__device__ __forceinline__ float bfus2f(ushort_t u) {
  union { unsigned int i; float f; } x; x.i = ((unsigned int)u) << 16; return x.f;
}
__device__ __forceinline__ ushort_t f2bfus(float f) {
  union { float f; unsigned int i; } x; x.f = f;
  unsigned int r = x.i + 0x7fffu + ((x.i >> 16) & 1u);
  return (ushort_t)(r >> 16);
}
__device__ __forceinline__ bf16_t us2bf(ushort_t u) {
  union { ushort_t s; bf16_t b; } x; x.s = u; return x.b;
}

__device__ __forceinline__ void async_copy16B(const ushort_t* g, ushort_t* l) {
#if USE_ASYNC_LDS
  __builtin_amdgcn_global_load_async_to_lds_b128(
      (GLOBAL_AS v4i_b*)g, (LDS_AS v4i_b*)l, 0, 0);
#else
  *(ushort4*)l       = *(const ushort4*)g;
  *(ushort4*)(l + 4) = *(const ushort4*)(g + 4);
#endif
}
__device__ __forceinline__ void wait_async() {
#if USE_ASYNC_LDS
  __builtin_amdgcn_s_wait_asynccnt(0);
#endif
}

// =====================================================================
// Batched WMMA GEMM:  C = act( (A@B + bias) * constScale * (*scalePtr) )
//   A: [M,K], f32 (A_BF16=0) or bf16 (A_BF16=1), row stride lda
//   B: bf16 stored K-CONTIGUOUS:  B[k][n] = Bu[n*ldb + k]  (B^T row-major)
//   C: f32/bf16; OUT_TRANS=1 stores C^T (bf16): Cp[n*ldc + m]
//   batch z: offX = (z/div)*sXo + (z%div)*sXi (elements)
// Block = 128 threads (4 waves), tile 64(M)x64(N), wave strip 16x64.
// Ping-pong LDS: B tile staged (async on CDNA5) in fragment-major layout so
// every fragment read is 2x ds_load_b128; 4 WMMAs issue back-to-back.
// =====================================================================
template <int A_BF16, int OUT_BF16, int OUT_TRANS, int ACT_RELU>
__global__ void __launch_bounds__(128)
wmma_gemm_t(
    const void* __restrict__ Ap, const void* __restrict__ Bp, void* __restrict__ Cp,
    const float* __restrict__ bias, const float* __restrict__ scalePtr, float constScale,
    int M, int N, int K, int lda, int ldb, int ldc,
    long long sAo, long long sAi, long long sBo, long long sBi,
    long long sCo, long long sCi, int div_)
{
  __shared__ ushort_t lA[2][4][16 * 32];   // per-wave A tile [m][k], ping-pong
  __shared__ ushort_t lB[2][4 * 32 * 16];  // fragment-major [tile][lane][e], ping-pong

  const int z  = blockIdx.z;
  const int zo = z / div_;
  const int zi = z - zo * div_;
  const float*    Af = ((const float*)Ap)    + (long long)zo * sAo + (long long)zi * sAi;
  const ushort_t* Au = ((const ushort_t*)Ap) + (long long)zo * sAo + (long long)zi * sAi;
  const ushort_t* Bu = ((const ushort_t*)Bp) + (long long)zo * sBo + (long long)zi * sBi;
  const long long offC = (long long)zo * sCo + (long long)zi * sCi;

  const int tid   = threadIdx.x;
  const int lane  = tid & 31;
  const int wv    = tid >> 5;
  const int r     = lane & 15;
  const int s     = lane >> 4;
  const int mBase = blockIdx.y * 64 + wv * 16;
  const int nBase = blockIdx.x * 64;
  const bool doCompute = (mBase < M);
  const bool nFull = (nBase + 64 <= N);
  const bool mFull = (mBase + 16 <= M);

  // ---- staging helpers ----
  auto stageB = [&](int k0, int buf) {
    const bool fullK = (k0 + 32 <= K);
    #pragma unroll
    for (int rr = 0; rr < 2; ++rr) {
      int runIdx = tid + rr * 128;            // = t*64 + lane2*2 + half
      int t     = runIdx >> 6;
      int lane2 = (runIdx >> 1) & 31;
      int half  = runIdx & 1;
      int r2 = lane2 & 15, s2 = lane2 >> 4;
      int n  = nBase + t * 16 + r2;
      int kb = k0 + half * 16 + s2 * 8;
      int dst = (t * 32 + lane2) * 16 + half * 8;
      if (fullK && (nFull || n < N)) {
        async_copy16B(Bu + (size_t)n * ldb + kb, &lB[buf][dst]);
      } else {
        #pragma unroll
        for (int i = 0; i < 8; ++i) {
          int k = kb + i;
          lB[buf][dst + i] = (k < K && n < N) ? Bu[(size_t)n * ldb + k] : (ushort_t)0;
        }
      }
    }
  };
  auto stageA = [&](int k0, int buf) {
    if (!doCompute) return;
    const bool fullK = (k0 + 32 <= K);
    #pragma unroll
    for (int rr = 0; rr < 2; ++rr) {
      int runIdx = lane + rr * 32;
      int mm = runIdx >> 2, q = runIdx & 3;
      int m = mBase + mm, kb = k0 + q * 8;
      int dst = mm * 32 + q * 8;
      if (fullK && (mFull || m < M)) {
        if (A_BF16) {
          async_copy16B(Au + (size_t)m * lda + kb, &lA[buf][wv][dst]);
        } else {
          const float* src = Af + (size_t)m * lda + kb;
          float4 f0 = *(const float4*)src;
          float4 f1 = *(const float4*)(src + 4);
          *(ushort4*)&lA[buf][wv][dst] =
              make_ushort4(f2bfus(f0.x), f2bfus(f0.y), f2bfus(f0.z), f2bfus(f0.w));
          *(ushort4*)&lA[buf][wv][dst + 4] =
              make_ushort4(f2bfus(f1.x), f2bfus(f1.y), f2bfus(f1.z), f2bfus(f1.w));
        }
      } else {
        #pragma unroll
        for (int i = 0; i < 8; ++i) {
          int k = kb + i;
          float v = 0.0f;
          if (m < M && k < K)
            v = A_BF16 ? bfus2f(Au[(size_t)m * lda + k]) : Af[(size_t)m * lda + k];
          lA[buf][wv][dst + i] = f2bfus(v);
        }
      }
    }
  };

  v8f acc[4];
  #pragma unroll
  for (int t = 0; t < 4; ++t)
    #pragma unroll
    for (int i = 0; i < 8; ++i) acc[t][i] = 0.0f;

  // ---- prologue: stage tile 0 ----
  stageB(0, 0);
  stageA(0, 0);
  wait_async();
  __syncthreads();

  const int nIter = (K + 31) / 32;
  for (int it = 0; it < nIter; ++it) {
    const int cur = it & 1, nxt = cur ^ 1;
    // stage next K tile (async: wave continues into compute immediately)
    if (it + 1 < nIter) {
      stageB((it + 1) * 32, nxt);
      stageA((it + 1) * 32, nxt);
    }
    if (doCompute) {
      // A fragment: lane = row M=r, two contiguous 8-elem K runs -> ds_load_b128 x2
      v16bf a;
      #pragma unroll
      for (int e = 0; e < 16; ++e) {
        int kk = ((e & 8) << 1) + s * 8 + (e & 7);
        a[e] = us2bf(lA[cur][wv][r * 32 + kk]);
      }
      // all four B fragments first, then 4 back-to-back WMMAs
      v16bf bfr[4];
      #pragma unroll
      for (int t = 0; t < 4; ++t)
        #pragma unroll
        for (int e = 0; e < 16; ++e) bfr[t][e] = us2bf(lB[cur][(t * 32 + lane) * 16 + e]);
      #pragma unroll
      for (int t = 0; t < 4; ++t)
        acc[t] = __builtin_amdgcn_wmma_f32_16x16x32_bf16(
            false, a, false, bfr[t], (short)0, acc[t], false, false);
    }
    wait_async();
    __syncthreads();
  }

  if (!doCompute) return;
  const float scl = constScale * (scalePtr ? *scalePtr : 1.0f);
  if (!OUT_TRANS) {
    // C layout: lane<16 -> N=r, M=i ; lane>=16 -> N=r, M=8+i
    #pragma unroll
    for (int t = 0; t < 4; ++t) {
      int n = nBase + t * 16 + r;
      #pragma unroll
      for (int i = 0; i < 8; ++i) {
        int m = mBase + s * 8 + i;
        if (m < M && n < N) {
          float v = acc[t][i];
          if (bias) v += bias[n];
          v *= scl;
          if (ACT_RELU && v < 0.0f) v = 0.0f;
          size_t ci = (size_t)offC + (size_t)m * ldc + n;
          if (OUT_BF16) ((ushort_t*)Cp)[ci] = f2bfus(v);
          else          ((float*)Cp)[ci]    = v;
        }
      }
    }
  } else {
    // transposed bf16 store: Cp[n*ldc + m]; 8 consecutive m per lane -> 16B store
    #pragma unroll
    for (int t = 0; t < 4; ++t) {
      int n = nBase + t * 16 + r;
      if (n >= N) continue;
      float vv[8];
      #pragma unroll
      for (int i = 0; i < 8; ++i) {
        float v = acc[t][i];
        if (bias) v += bias[n];
        v *= scl;
        if (ACT_RELU && v < 0.0f) v = 0.0f;
        vv[i] = v;
      }
      int m0 = mBase + s * 8;
      ushort_t* cp = (ushort_t*)Cp + (size_t)offC + (size_t)n * ldc + m0;
      if (m0 + 8 <= M) {
        *(ushort4*)cp       = make_ushort4(f2bfus(vv[0]), f2bfus(vv[1]), f2bfus(vv[2]), f2bfus(vv[3]));
        *(ushort4*)(cp + 4) = make_ushort4(f2bfus(vv[4]), f2bfus(vv[5]), f2bfus(vv[6]), f2bfus(vv[7]));
      } else {
        for (int i = 0; i < 8; ++i) if (m0 + i < M) cp[i] = f2bfus(vv[i]);
      }
    }
  }
}

// =====================================================================
// Small support kernels
// =====================================================================
__global__ void conv1d_bf(const float* __restrict__ src, ushort_t* __restrict__ dst, int count) {
  int i = blockIdx.x * blockDim.x + threadIdx.x;
  if (i < count) dst[i] = f2bfus(src[i]);
}

// src [R,C] -> dst [C,R] bf16  (weights stored K-contiguous for the GEMM)
__global__ void convT_bf(const float* __restrict__ src, ushort_t* __restrict__ dst, int R, int C) {
  int i = blockIdx.x * blockDim.x + threadIdx.x;
  if (i < R * C) {
    int c = i / R, rr = i - c * R;
    dst[i] = f2bfus(src[(size_t)rr * C + c]);
  }
}

__global__ void h_init_kernel(const float* __restrict__ h0, float* __restrict__ h,
                              ushort_t* __restrict__ hbf, int count) {
  int i = blockIdx.x * blockDim.x + threadIdx.x;
  if (i < count) { float v = h0[i]; h[i] = v; hbf[i] = f2bfus(v); }
}

__global__ void sum_small_kernel(const float* __restrict__ p, int n, float* __restrict__ out) {
  int l = threadIdx.x;
  float v = (l < n) ? p[l] : 0.0f;
  for (int o = 16; o > 0; o >>= 1) v += __shfl_xor(v, o, 32);
  if (l == 0) *out = v;
}

__global__ void gru_pointwise_kernel(const float* __restrict__ wx, const float* __restrict__ wh,
                                     float* __restrict__ h, ushort_t* __restrict__ hbf,
                                     ushort_t* __restrict__ qemb_bf, int t) {
  int idx = blockIdx.x * blockDim.x + threadIdx.x;
  if (idx >= 64 * 768) return;
  int b = idx / 768, d = idx - b * 768;
  const float* wxb = wx + (size_t)b * 2304;
  const float* whb = wh + (size_t)b * 2304;
  float rg = 1.0f / (1.0f + __expf(-(wxb[d] + whb[d])));
  float zg = 1.0f / (1.0f + __expf(-(wxb[768 + d] + whb[768 + d])));
  float ng = tanhf(wxb[1536 + d] + rg * whb[1536 + d]);
  float hp = h[idx];
  float hn = (1.0f - zg) * ng + zg * hp;
  h[idx] = hn;
  hbf[idx] = f2bfus(hn);
  qemb_bf[(size_t)b * (32 * 768) + (size_t)t * 768 + d] = f2bfus(hn);
}

__global__ void softmax_rows_kernel(const float* __restrict__ S, ushort_t* __restrict__ P, int cols) {
  int row = blockIdx.x;
  const float* sr = S + (size_t)row * cols;
  ushort_t* pr = P + (size_t)row * cols;
  int lane = threadIdx.x;
  float mx = -3.4e38f;
  for (int c = lane; c < cols; c += 32) mx = fmaxf(mx, sr[c]);
  for (int o = 16; o > 0; o >>= 1) mx = fmaxf(mx, __shfl_xor(mx, o, 32));
  float sum = 0.0f;
  for (int c = lane; c < cols; c += 32) sum += __expf(sr[c] - mx);
  for (int o = 16; o > 0; o >>= 1) sum += __shfl_xor(sum, o, 32);
  float inv = 1.0f / sum;
  for (int c = lane; c < cols; c += 32) pr[c] = f2bfus(__expf(sr[c] - mx) * inv);
}

__global__ void pool_q_kernel(const float* __restrict__ out, const float* __restrict__ pool,
                              float* __restrict__ dst) {
  int idx = blockIdx.x * blockDim.x + threadIdx.x;
  if (idx >= 64 * 768) return;
  int b = idx / 768, d = idx - b * 768;
  float s = 0.0f;
  for (int q = 0; q < 32; ++q) s += pool[q] * out[((size_t)b * 32 + q) * 768 + d];
  dst[idx] = s;
}

__global__ void concat_bf_kernel(const float* __restrict__ a, const float* __restrict__ b,
                                 ushort_t* __restrict__ dst) {
  int idx = blockIdx.x * blockDim.x + threadIdx.x;
  if (idx >= 64 * 1536) return;
  int r = idx / 1536, c = idx - r * 1536;
  float v = (c < 768) ? a[(size_t)r * 768 + c] : b[(size_t)r * 768 + (c - 768)];
  dst[idx] = f2bfus(v);
}

// =====================================================================
// Host side
// =====================================================================
static inline void launch_gemm(hipStream_t st,
    const void* A, int aBf16, long long sAo, long long sAi, int lda,
    const ushort_t* B, long long sBo, long long sBi, int ldb,
    void* C, int outBf16, int outTrans, long long sCo, long long sCi, int ldc,
    const float* bias, const float* scalePtr, float constScale, int actRelu,
    int M, int N, int K, int batch, int div_)
{
  dim3 g((N + 63) / 64, (M + 63) / 64, batch), b(128);
  #define GEMM_ARGS A, B, C, bias, scalePtr, constScale, M, N, K, lda, ldb, ldc, \
                    sAo, sAi, sBo, sBi, sCo, sCi, div_
  if (outTrans)                          wmma_gemm_t<0,1,1,0><<<g, b, 0, st>>>(GEMM_ARGS);
  else if (!aBf16 && outBf16)            wmma_gemm_t<0,1,0,0><<<g, b, 0, st>>>(GEMM_ARGS);
  else if (aBf16 && outBf16 && actRelu)  wmma_gemm_t<1,1,0,1><<<g, b, 0, st>>>(GEMM_ARGS);
  else if (aBf16 && outBf16)             wmma_gemm_t<1,1,0,0><<<g, b, 0, st>>>(GEMM_ARGS);
  else                                   wmma_gemm_t<1,0,0,0><<<g, b, 0, st>>>(GEMM_ARGS);
  #undef GEMM_ARGS
}

extern "C" void kernel_launch(void* const* d_in, const int* in_sizes, int n_in,
                              void* d_out, int out_size, void* d_ws, size_t ws_size,
                              hipStream_t stream) {
  (void)in_sizes; (void)n_in; (void)out_size; (void)ws_size;
  const int D = 768, B = 64, T = 32, NL = 900, H3 = 2304, DK = 384;
  const int LDVT = 904;  // padded ld for transposed V (16B alignment)

  const float* img    = (const float*)d_in[1];   // [64,901,768]
  const float* h0     = (const float*)d_in[2];
  const float* w_ih   = (const float*)d_in[3];   // [2304,768] == already K-contiguous B
  const float* w_hh   = (const float*)d_in[4];
  const float* b_ih   = (const float*)d_in[5];
  const float* b_hh   = (const float*)d_in[6];
  const float* ga_w   = (const float*)d_in[7];
  const float* ga_b   = (const float*)d_in[8];
  const float* ga_pool= (const float*)d_in[9];
  const float* la_w   = (const float*)d_in[10];
  const float* la_b   = (const float*)d_in[11];
  const float* la_pool= (const float*)d_in[12];
  const float* go_w   = (const float*)d_in[13];
  const float* go_b   = (const float*)d_in[14];
  const float* go_pool= (const float*)d_in[15];
  const float* f1_w   = (const float*)d_in[16];
  const float* f1_b   = (const float*)d_in[17];
  const float* f2_w   = (const float*)d_in[18];
  const float* f2_b   = (const float*)d_in[19];
  const float* f3_w   = (const float*)d_in[20];
  const float* f3_b   = (const float*)d_in[21];

  char* wsp = (char*)d_ws;
  size_t off = 0;
  auto alloc = [&](size_t bytes) -> void* {
    void* p = wsp + off;
    off = (off + bytes + 255) & ~(size_t)255;
    return p;
  };
  const size_t DD = (size_t)D * D;

  ushort_t* w_ga2  = (ushort_t*)alloc(DD * 2);   // transposed [N,K]
  ushort_t* w_ga3  = (ushort_t*)alloc(DD * 2);
  ushort_t* w_go2  = (ushort_t*)alloc(DD * 2);
  ushort_t* w_go3  = (ushort_t*)alloc(DD * 2);
  ushort_t* w_la0  = (ushort_t*)alloc(DD * 2);
  ushort_t* w_la1  = (ushort_t*)alloc(DD * 2);
  ushort_t* w_la2  = (ushort_t*)alloc(DD * 2);
  ushort_t* w_la3  = (ushort_t*)alloc(DD * 2);
  ushort_t* w_ihB  = (ushort_t*)alloc((size_t)D * H3 * 2);  // [2304,768] bf16
  ushort_t* w_hhB  = (ushort_t*)alloc((size_t)D * H3 * 2);
  ushort_t* w_f1   = (ushort_t*)alloc((size_t)1536 * 1024 * 2); // [1024,1536]
  ushort_t* w_f2   = (ushort_t*)alloc((size_t)1024 * 512 * 2);  // [512,1024]
  ushort_t* w_f3   = (ushort_t*)alloc((size_t)512 * 1024 * 2);  // [1024,512]
  ushort_t* t0_bf  = (ushort_t*)alloc((size_t)B * D * 2);
  ushort_t* t0g_bf = (ushort_t*)alloc((size_t)B * D * 2);
  ushort_t* a_bf   = (ushort_t*)alloc((size_t)B * D * 2);
  ushort_t* h_bf   = (ushort_t*)alloc((size_t)B * D * 2);
  ushort_t* qemb_bf= (ushort_t*)alloc((size_t)B * T * D * 2);
  ushort_t* Kla_bf = (ushort_t*)alloc((size_t)B * NL * D * 2);       // [b][n][k]
  ushort_t* Vt_bf  = (ushort_t*)alloc((size_t)B * D * LDVT * 2);     // [b][dcol][key], ld=904
  ushort_t* Q_bf   = (ushort_t*)alloc((size_t)B * T * D * 2);
  ushort_t* P_bf   = (ushort_t*)alloc((size_t)B * 2 * T * NL * 2);
  ushort_t* ctx_bf = (ushort_t*)alloc((size_t)B * T * D * 2);
  ushort_t* x0_bf  = (ushort_t*)alloc((size_t)B * 1536 * 2);
  ushort_t* x1_bf  = (ushort_t*)alloc((size_t)B * 1024 * 2);
  ushort_t* x2_bf  = (ushort_t*)alloc((size_t)B * 512 * 2);
  float* wx        = (float*)alloc((size_t)B * H3 * 4);
  float* wh        = (float*)alloc((size_t)B * H3 * 4);
  float* h_f       = (float*)alloc((size_t)B * D * 4);
  float* scores    = (float*)alloc((size_t)B * 2 * T * NL * 4);
  float* outla     = (float*)alloc((size_t)B * T * D * 4);
  float* local_out = (float*)alloc((size_t)B * D * 4);
  float* global_out= (float*)alloc((size_t)B * D * 4);
  float* go_sum    = (float*)alloc(256);

  const int CB = 256;
  auto nb = [](int n, int b) { return (n + b - 1) / b; };

  // ---- weight conversions ----
  convT_bf<<<nb((int)DD, CB), CB, 0, stream>>>(ga_w + 2 * DD, w_ga2, D, D);
  convT_bf<<<nb((int)DD, CB), CB, 0, stream>>>(ga_w + 3 * DD, w_ga3, D, D);
  convT_bf<<<nb((int)DD, CB), CB, 0, stream>>>(go_w + 2 * DD, w_go2, D, D);
  convT_bf<<<nb((int)DD, CB), CB, 0, stream>>>(go_w + 3 * DD, w_go3, D, D);
  convT_bf<<<nb((int)DD, CB), CB, 0, stream>>>(la_w + 0 * DD, w_la0, D, D);
  convT_bf<<<nb((int)DD, CB), CB, 0, stream>>>(la_w + 1 * DD, w_la1, D, D);
  convT_bf<<<nb((int)DD, CB), CB, 0, stream>>>(la_w + 2 * DD, w_la2, D, D);
  convT_bf<<<nb((int)DD, CB), CB, 0, stream>>>(la_w + 3 * DD, w_la3, D, D);
  conv1d_bf<<<nb(D * H3, CB), CB, 0, stream>>>(w_ih, w_ihB, D * H3);   // already [N,K]
  conv1d_bf<<<nb(D * H3, CB), CB, 0, stream>>>(w_hh, w_hhB, D * H3);
  convT_bf<<<nb(1536 * 1024, CB), CB, 0, stream>>>(f1_w, w_f1, 1536, 1024);
  convT_bf<<<nb(1024 * 512, CB), CB, 0, stream>>>(f2_w, w_f2, 1024, 512);
  convT_bf<<<nb(512 * 1024, CB), CB, 0, stream>>>(f3_w, w_f3, 512, 1024);

  h_init_kernel<<<nb(B * D, CB), CB, 0, stream>>>(h0, h_f, h_bf, B * D);
  sum_small_kernel<<<1, 32, 0, stream>>>(go_pool, 32, go_sum);

  // ---- global-attention constants (softmax over len-1 keys == identity) ----
  launch_gemm(stream, img, 0, 0, 0, 901 * D, w_ga2, 0, 0, D,
              t0_bf, 1, 0, 0, 0, D, ga_b + 2 * D, nullptr, 1.0f, 0, B, D, D, 1, 1);
  launch_gemm(stream, t0_bf, 1, 0, 0, D, w_ga3, 0, 0, D,
              a_bf, 1, 0, 0, 0, D, ga_b + 3 * D, ga_pool, 1.0f, 0, B, D, D, 1, 1);
  launch_gemm(stream, img, 0, 0, 0, 901 * D, w_go2, 0, 0, D,
              t0g_bf, 1, 0, 0, 0, D, go_b + 2 * D, nullptr, 1.0f, 0, B, D, D, 1, 1);
  launch_gemm(stream, t0g_bf, 1, 0, 0, D, w_go3, 0, 0, D,
              global_out, 0, 0, 0, 0, D, go_b + 3 * D, go_sum, 1.0f, 0, B, D, D, 1, 1);

  // ---- GRU: wx constant across steps ----
  launch_gemm(stream, a_bf, 1, 0, 0, D, w_ihB, 0, 0, D,
              wx, 0, 0, 0, 0, H3, b_ih, nullptr, 1.0f, 0, B, H3, D, 1, 1);
  for (int t = 0; t < T; ++t) {
    launch_gemm(stream, h_bf, 1, 0, 0, D, w_hhB, 0, 0, D,
                wh, 0, 0, 0, 0, H3, b_hh, nullptr, 1.0f, 0, B, H3, D, 1, 1);
    gru_pointwise_kernel<<<nb(B * D, CB), CB, 0, stream>>>(wx, wh, h_f, h_bf, qemb_bf, t);
  }

  // ---- local attention K/V projections (dominant GEMMs) ----
  launch_gemm(stream, img + D, 0, (long long)901 * D, 0, D, w_la1, 0, 0, D,
              Kla_bf, 1, 0, (long long)NL * D, 0, D, la_b + 1 * D, nullptr, 1.0f, 0,
              NL, D, D, B, 1);
  // V written transposed (per batch: [dcol][key], ld=904) for K-contiguous ctx GEMM
  launch_gemm(stream, img + D, 0, (long long)901 * D, 0, D, w_la2, 0, 0, D,
              Vt_bf, 1, 1, (long long)D * LDVT, 0, LDVT, la_b + 2 * D, nullptr, 1.0f, 0,
              NL, D, D, B, 1);
  // Q projection
  launch_gemm(stream, qemb_bf, 1, 0, 0, D, w_la0, 0, 0, D,
              Q_bf, 1, 0, 0, 0, D, la_b + 0 * D, nullptr, 1.0f, 0, B * T, D, D, 1, 1);
  // scores[b,h] = Q[b,h] @ K[b,h]^T / sqrt(dk); B operand = Kla (K-contiguous)
  launch_gemm(stream, Q_bf, 1, (long long)T * D, DK, D,
              Kla_bf, (long long)NL * D, DK, D,
              scores, 0, 0, (long long)2 * T * NL, (long long)T * NL, NL,
              nullptr, nullptr, 0.05103103630798287f /* 1/sqrt(384) */, 0,
              T, NL, DK, B * 2, 2);
  softmax_rows_kernel<<<B * 2 * T, 32, 0, stream>>>(scores, P_bf, NL);
  // ctx[b,h] = P[b,h] @ V[b,h]; B operand = Vt (K-contiguous over keys)
  launch_gemm(stream, P_bf, 1, (long long)2 * T * NL, (long long)T * NL, NL,
              Vt_bf, (long long)D * LDVT, (long long)DK * LDVT, LDVT,
              ctx_bf, 1, 0, (long long)T * D, DK, D,
              nullptr, nullptr, 1.0f, 0, T, DK, NL, B * 2, 2);
  // output projection + query pooling
  launch_gemm(stream, ctx_bf, 1, 0, 0, D, w_la3, 0, 0, D,
              outla, 0, 0, 0, 0, D, la_b + 3 * D, nullptr, 1.0f, 0, B * T, D, D, 1, 1);
  pool_q_kernel<<<nb(B * D, CB), CB, 0, stream>>>(outla, la_pool, local_out);

  // ---- FC head ----
  concat_bf_kernel<<<nb(B * 1536, CB), CB, 0, stream>>>(local_out, global_out, x0_bf);
  launch_gemm(stream, x0_bf, 1, 0, 0, 1536, w_f1, 0, 0, 1536,
              x1_bf, 1, 0, 0, 0, 1024, f1_b, nullptr, 1.0f, 0, B, 1024, 1536, 1, 1);
  launch_gemm(stream, x1_bf, 1, 0, 0, 1024, w_f2, 0, 0, 1024,
              x2_bf, 1, 0, 0, 0, 512, f2_b, nullptr, 1.0f, 1, B, 512, 1024, 1, 1);
  launch_gemm(stream, x2_bf, 1, 0, 0, 512, w_f3, 0, 0, 512,
              d_out, 0, 0, 0, 0, 1024, f3_b, nullptr, 1.0f, 0, B, 1024, 512, 1, 1);
}
